// MultiHeadAttention_56298431316055
// MI455X (gfx1250) — matrix operand order repaired and checked by reading, not compiled, not run on hardware
//
#include <hip/hip_runtime.h>
#include <hip/hip_bf16.h>

// ---------------------------------------------------------------------------
// MI455X (gfx1250) multi-head causal attention, bf16 WMMA path, round 3.
//  - All LDS tiles stored in WMMA fragment order ("swizzled") so every
//    fragment load is 2x ds_load_b128 instead of 16x ds_load_u16.
//  - bf16 conversion via v_cvt_pk_bf16_f32 when available.
//  - Async global->LDS copy (global_load_async_to_lds_b128) for the K tile,
//    with exact builtin parameter types per the round-2 diagnostic.
// ---------------------------------------------------------------------------

typedef __attribute__((ext_vector_type(16))) __bf16 v16bf;
typedef __attribute__((ext_vector_type(2)))  __bf16 v2bf;
typedef __attribute__((ext_vector_type(8)))  float  v8f;

#if __has_builtin(__builtin_amdgcn_global_load_async_to_lds_b128) && \
    __has_builtin(__builtin_amdgcn_s_wait_asynccnt)
#define USE_ASYNC_COPY 1
typedef int vi4 __attribute__((vector_size(16)));
typedef __attribute__((address_space(1))) vi4 gvi4;   // global int4 vector
typedef __attribute__((address_space(3))) vi4 lvi4;   // LDS int4 vector
#else
#define USE_ASYNC_COPY 0
#endif

// pack two f32 -> two bf16 (RNE)
__device__ __forceinline__ unsigned int pkbf(float a, float b) {
#if __has_builtin(__builtin_amdgcn_cvt_pk_bf16_f32)
  union { v2bf v; unsigned int u; } x;
  x.v = __builtin_amdgcn_cvt_pk_bf16_f32(a, b);
  return x.u;
#else
  unsigned int ua = __float_as_uint(a), ub = __float_as_uint(b);
  ua += 0x7FFFu + ((ua >> 16) & 1u);
  ub += 0x7FFFu + ((ub >> 16) & 1u);
  return (ua >> 16) | (ub & 0xFFFF0000u);
#endif
}
__device__ __forceinline__ unsigned short bf1(float a) {
  return (unsigned short)pkbf(a, a);
}

// A-fragment layout within a K=32 slice: element (row,k) stored at
// row*stride + swzA(k) = k (contiguous, matches B frags).
__device__ __forceinline__ int swzA(int k) {
  return k;
}

// Load one 16-element bf16 fragment (32 contiguous bytes) -> 2x ds_load_b128.
union FragU { uint4 q[2]; v16bf v; };
__device__ __forceinline__ v16bf ld_frag(const unsigned short* s, int e0) {
  FragU f;
  f.q[0] = *(const uint4*)(s + e0);
  f.q[1] = *(const uint4*)(s + e0 + 8);
  return f.v;
}

#define WMMA_BF16(a, b, c) \
  __builtin_amdgcn_wmma_f32_16x16x32_bf16(false, (a), false, (b), (short)0, (c), false, false)

// ---------------------------------------------------------------------------
// Kernel 1: QKV projection.  Block = 128 rows x 64 cols x K=1024 GEMM,
// 8 waves, each wave: 16 rows x 64 cols -> 4 WMMAs per K-step.
// z in {0,1,2} selects q/k/v.  Output bf16 [B,H,S,dk].
// ---------------------------------------------------------------------------
__global__ __launch_bounds__(256) void qkv_proj_kernel(
    const float* __restrict__ queries, const float* __restrict__ keys,
    const float* __restrict__ values,
    const float* __restrict__ Wq, const float* __restrict__ bq,
    const float* __restrict__ Wk, const float* __restrict__ bk,
    const float* __restrict__ Wv, const float* __restrict__ bv,
    unsigned short* __restrict__ qkv_ws) {
  constexpr int S = 2048, D = 1024, H = 16, DK = 64;
  const size_t NQ = (size_t)4 * H * S * DK;

  __shared__ __align__(16) unsigned short sA[128 * 40];  // A frags, swizzled
  __shared__ __align__(16) unsigned short sB[64 * 40];   // B frags: col-major

  const int z = blockIdx.z;
  const float* in   = (z == 0) ? queries : (z == 1) ? keys : values;
  const float* W    = (z == 0) ? Wq : (z == 1) ? Wk : Wv;
  const float* bias = (z == 0) ? bq : (z == 1) ? bk : bv;
  unsigned short* out = qkv_ws + (size_t)z * NQ;

  const int bh = blockIdx.y;                 // b*H + h
  const int b  = bh / H, h = bh % H;
  const int s0 = blockIdx.x * 128;
  const int t = threadIdx.x, lane = t & 31, w = t >> 5;
  const int jc = lane & 15, hh = lane >> 4;

  const float* Ain = in + ((size_t)b * S + s0) * D;
  const float* Bw  = W + (size_t)h * D * DK;

  v8f c0 = {}, c1 = {}, c2 = {}, c3 = {};

  for (int kk = 0; kk < D; kk += 32) {
    // A tile 128x32: f32 float4 loads -> packed bf16, swizzled store
#pragma unroll
    for (int p = 0; p < 4; ++p) {
      const int idx = t + p * 256;                 // 1024 float4s
      const int r = idx >> 3, c4 = (idx & 7) * 4;
      const float4 v = *(const float4*)(Ain + (size_t)r * D + kk + c4);
      uint2 u; u.x = pkbf(v.x, v.y); u.y = pkbf(v.z, v.w);
      *(uint2*)&sA[r * 40 + swzA(c4)] = u;
    }
    // B tile 32x64: k-strided f32 loads (lane-coalesced in n), column-major
    {
      const int n  = t & 63;
      const int kq = (t >> 6) * 4;
#pragma unroll
      for (int p = 0; p < 2; ++p) {
        const int k0 = kq + p * 16;
        const float* src = Bw + (size_t)(kk + k0) * DK + n;
        uint2 u;
        u.x = pkbf(src[0], src[DK]);
        u.y = pkbf(src[2 * DK], src[3 * DK]);
        *(uint2*)&sB[n * 40 + k0] = u;
      }
    }
    if (kk + 32 < D)
      __builtin_prefetch(Ain + (size_t)(t >> 1) * D + kk + 32, 0, 1);
    __syncthreads();

    const v16bf a  = ld_frag(sA, (w * 16 + jc) * 40 + hh * 16);
    const v16bf b0 = ld_frag(sB, (jc)*40      + hh * 16);
    const v16bf b1 = ld_frag(sB, (16 + jc) * 40 + hh * 16);
    const v16bf b2 = ld_frag(sB, (32 + jc) * 40 + hh * 16);
    const v16bf b3 = ld_frag(sB, (48 + jc) * 40 + hh * 16);
    c0 = WMMA_BF16(a, b0, c0);
    c1 = WMMA_BF16(a, b1, c1);
    c2 = WMMA_BF16(a, b2, c2);
    c3 = WMMA_BF16(a, b3, c3);
    __syncthreads();
  }

  // bias + bf16 store to [B,H,S,dk]
  unsigned short* orow = out + ((size_t)bh * S + s0) * DK;
  const float bb0 = bias[h * DK + jc];
  const float bb1 = bias[h * DK + 16 + jc];
  const float bb2 = bias[h * DK + 32 + jc];
  const float bb3 = bias[h * DK + 48 + jc];
#pragma unroll
  for (int r = 0; r < 8; ++r) {
    const int row = w * 16 + r + hh * 8;
    unsigned short* dst = orow + (size_t)row * DK + jc;
    dst[0]  = bf1(c0[r] + bb0);
    dst[16] = bf1(c1[r] + bb1);
    dst[32] = bf1(c2[r] + bb2);
    dst[48] = bf1(c3[r] + bb3);
  }
}

// ---------------------------------------------------------------------------
// Kernel 2: flash attention.  Block = 128 query rows of one (b,h); each wave
// holds a 16-row Q tile as 2 A-frags.  Stream 32-key K/V tiles via LDS.
// 8 WMMAs / wave / key-tile + online softmax.
// ---------------------------------------------------------------------------
__global__ __launch_bounds__(256) void flash_attn_kernel(
    const unsigned short* __restrict__ qkv, unsigned short* __restrict__ ctx) {
  constexpr int S = 2048, H = 16, DK = 64;
  constexpr float SCALE = 0.125f;            // 1/sqrt(64)
  const size_t NQ = (size_t)4 * H * S * DK;

  __shared__ __align__(16) unsigned short sQ[128 * 72];  // Q, A-frag order
  __shared__ __align__(16) unsigned short sK[32 * 72];   // K, row-major
  __shared__ __align__(16) unsigned short sVt[64 * 40];  // V^T, B-frag order
  __shared__ __align__(16) unsigned short sP[8 * 16 * 40];

  const unsigned short* qbase = qkv;
  const unsigned short* kbase = qkv + NQ;
  const unsigned short* vbase = qkv + 2 * NQ;

  const int bh = blockIdx.y;
  const int b  = bh / H, h = bh % H;
  const int q0 = blockIdx.x * 128;
  const int t = threadIdx.x, lane = t & 31, w = t >> 5;
  const int jc = lane & 15, hh = lane >> 4, hi8 = hh * 8;
  unsigned short* sPw = &sP[w * 16 * 40];
  const int spo = jc;   // swzA(jc), jc<16

  const unsigned short* Qg = qbase + ((size_t)bh * S + q0) * DK;
  const unsigned short* Kg = kbase + (size_t)bh * S * DK;
  const unsigned short* Vg = vbase + (size_t)bh * S * DK;

  // stage Q 128x64 in A-fragment order (row-major, k contiguous, chunks unpermuted)
#pragma unroll
  for (int p = 0; p < 4; ++p) {
    const int idx = t + p * 256;
    const int r = idx >> 3, c8 = (idx & 7) * 8;
    const uint4 v = *(const uint4*)(Qg + (size_t)r * DK + c8);
    *(uint4*)&sQ[r * 72 + (c8 & 32) + swzA(c8 & 31)] = v;
  }
  __syncthreads();

  const v16bf qa0 = ld_frag(sQ, (w * 16 + jc) * 72 + hh * 16);
  const v16bf qa1 = ld_frag(sQ, (w * 16 + jc) * 72 + 32 + hh * 16);

  v8f acc0 = {}, acc1 = {}, acc2 = {}, acc3 = {};
  float mrun[8], lrun[8];
#pragma unroll
  for (int r = 0; r < 8; ++r) { mrun[r] = -1e30f; lrun[r] = 0.0f; }

  const int myq = q0 + w * 16;
  const int nkt = (q0 + 128) / 32;           // causal bound for this block

  for (int kt = 0; kt < nkt; ++kt) {
    const int k0g = kt * 32;
    // stage K tile (pure copy -> async DMA to LDS)
    {
      const int r = t >> 3, c8 = (t & 7) * 8;
      const unsigned short* gsrc = Kg + (size_t)(k0g + r) * DK + c8;
      unsigned short* ldst = &sK[r * 72 + c8];
#if USE_ASYNC_COPY
      __builtin_amdgcn_global_load_async_to_lds_b128(
          (gvi4*)gsrc, (lvi4*)ldst, 0, 0);
#else
      *(uint4*)ldst = *(const uint4*)gsrc;
#endif
    }
    // stage V transposed into B-fragment order (lane-coalesced u16 loads)
    {
      const int dk = t & 63;
      const int kq = (t >> 6) * 4;
#pragma unroll
      for (int p = 0; p < 2; ++p) {
        const int k0 = kq + p * 16;
        const unsigned short* src = Vg + (size_t)(k0g + k0) * DK + dk;
        uint2 u;
        u.x = (unsigned int)src[0]      | ((unsigned int)src[DK] << 16);
        u.y = (unsigned int)src[2 * DK] | ((unsigned int)src[3 * DK] << 16);
        *(uint2*)&sVt[dk * 40 + k0] = u;
      }
    }
#if USE_ASYNC_COPY
    __builtin_amdgcn_s_wait_asynccnt(0);
#endif
    __syncthreads();

    // scores 16x32 = Q(16x64) * K^T  (K rows are B-frag contiguous)
    v8f s0 = {}, s1 = {};
    {
      const v16bf k00 = ld_frag(sK, (jc)*72      + hh * 16);
      const v16bf k01 = ld_frag(sK, (jc)*72 + 32 + hh * 16);
      s0 = WMMA_BF16(qa0, k00, s0);
      s0 = WMMA_BF16(qa1, k01, s0);
      const v16bf k10 = ld_frag(sK, (16 + jc) * 72      + hh * 16);
      const v16bf k11 = ld_frag(sK, (16 + jc) * 72 + 32 + hh * 16);
      s1 = WMMA_BF16(qa0, k10, s1);
      s1 = WMMA_BF16(qa1, k11, s1);
    }

    // online softmax per owned row (lanes 0-15: rows 0..7, 16-31: 8..15)
#pragma unroll
    for (int r = 0; r < 8; ++r) {
      const int qrow = myq + r + hi8;
      float v0 = s0[r] * SCALE;
      float v1 = s1[r] * SCALE;
      if (k0g + jc      > qrow) v0 = -1e30f;
      if (k0g + 16 + jc > qrow) v1 = -1e30f;
      float tm = fmaxf(v0, v1);
#pragma unroll
      for (int m = 8; m >= 1; m >>= 1) tm = fmaxf(tm, __shfl_xor(tm, m, 32));
      const float newm = fmaxf(mrun[r], tm);
      const float cf   = __expf(mrun[r] - newm);
      mrun[r] = newm;
      const float p0 = (v0 > -1e29f) ? __expf(v0 - newm) : 0.0f;
      const float p1 = (v1 > -1e29f) ? __expf(v1 - newm) : 0.0f;
      float ps = p0 + p1;
#pragma unroll
      for (int m = 8; m >= 1; m >>= 1) ps += __shfl_xor(ps, m, 32);
      lrun[r] = lrun[r] * cf + ps;
      acc0[r] = acc0[r] * cf; acc1[r] = acc1[r] * cf;
      acc2[r] = acc2[r] * cf; acc3[r] = acc3[r] * cf;
      // store P directly in swizzled A-fragment order
      unsigned short* pd = &sPw[(r + hi8) * 40 + spo];
      pd[0] = bf1(p0);            // k = jc
      pd[16] = bf1(p1);           // k = jc + 16  ->  swzA + 16
    }

    // ctx += P(16x32) * V(32x64)   (V^T is B-frag contiguous)
    const v16bf pa  = ld_frag(sPw, jc * 40 + hh * 16);
    const v16bf vb0 = ld_frag(sVt, (jc)*40      + hh * 16);
    const v16bf vb1 = ld_frag(sVt, (16 + jc) * 40 + hh * 16);
    const v16bf vb2 = ld_frag(sVt, (32 + jc) * 40 + hh * 16);
    const v16bf vb3 = ld_frag(sVt, (48 + jc) * 40 + hh * 16);
    acc0 = WMMA_BF16(pa, vb0, acc0);
    acc1 = WMMA_BF16(pa, vb1, acc1);
    acc2 = WMMA_BF16(pa, vb2, acc2);
    acc3 = WMMA_BF16(pa, vb3, acc3);
    __syncthreads();
  }

  // normalize + store ctx bf16 as [B, S, H*dk]
#pragma unroll
  for (int r = 0; r < 8; ++r) {
    const int row = myq + r + hi8;
    const float inv = 1.0f / lrun[r];
    unsigned short* dst =
        ctx + ((size_t)b * S + row) * (H * DK) + h * DK + jc;
    dst[0]  = bf1(acc0[r] * inv);
    dst[16] = bf1(acc1[r] * inv);
    dst[32] = bf1(acc2[r] * inv);
    dst[48] = bf1(acc3[r] * inv);
  }
}

// ---------------------------------------------------------------------------
// Kernel 3: output projection.  out[8192,1024] = ctx @ Wo + bo, f32 out.
// Block = 128x64 tile; wave = 16 rows x 64 cols -> 4 WMMAs / K-step.
// ---------------------------------------------------------------------------
__global__ __launch_bounds__(256) void out_proj_kernel(
    const unsigned short* __restrict__ ctx, const float* __restrict__ Wo,
    const float* __restrict__ bo, float* __restrict__ out) {
  constexpr int D = 1024;
  __shared__ __align__(16) unsigned short sA[128 * 40];
  __shared__ __align__(16) unsigned short sB[64 * 40];

  const int row0 = blockIdx.x * 128;
  const int col0 = blockIdx.y * 64;
  const int t = threadIdx.x, lane = t & 31, w = t >> 5;
  const int jc = lane & 15, hh = lane >> 4;

  v8f c0 = {}, c1 = {}, c2 = {}, c3 = {};

  for (int kk = 0; kk < D; kk += 32) {
    // A tile 128x32 (already bf16): uint4 copy with swizzled placement
#pragma unroll
    for (int p = 0; p < 2; ++p) {
      const int idx = t + p * 256;                // 512 uint4s
      const int r = idx >> 2, c8 = (idx & 3) * 8;
      const uint4 v = *(const uint4*)(ctx + (size_t)(row0 + r) * D + kk + c8);
      *(uint4*)&sA[r * 40 + swzA(c8)] = v;
    }
    // B tile 32x64 from Wo (k-strided f32, lane-coalesced in n)
    {
      const int n  = t & 63;
      const int kq = (t >> 6) * 4;
#pragma unroll
      for (int p = 0; p < 2; ++p) {
        const int k0 = kq + p * 16;
        const float* src = Wo + (size_t)(kk + k0) * D + col0 + n;
        uint2 u;
        u.x = pkbf(src[0], src[D]);
        u.y = pkbf(src[2 * D], src[3 * D]);
        *(uint2*)&sB[n * 40 + k0] = u;
      }
    }
    if (kk + 32 < D)
      __builtin_prefetch(ctx + (size_t)(row0 + (t >> 1)) * D + kk + 32, 0, 1);
    __syncthreads();

    const v16bf a  = ld_frag(sA, (w * 16 + jc) * 40 + hh * 16);
    const v16bf b0 = ld_frag(sB, (jc)*40      + hh * 16);
    const v16bf b1 = ld_frag(sB, (16 + jc) * 40 + hh * 16);
    const v16bf b2 = ld_frag(sB, (32 + jc) * 40 + hh * 16);
    const v16bf b3 = ld_frag(sB, (48 + jc) * 40 + hh * 16);
    c0 = WMMA_BF16(a, b0, c0);
    c1 = WMMA_BF16(a, b1, c1);
    c2 = WMMA_BF16(a, b2, c2);
    c3 = WMMA_BF16(a, b3, c3);
    __syncthreads();
  }

  const int cA = col0 + jc;
  const float b0v = bo[cA], b1v = bo[cA + 16], b2v = bo[cA + 32], b3v = bo[cA + 48];
#pragma unroll
  for (int r = 0; r < 8; ++r) {
    const int row = row0 + w * 16 + r + hh * 8;
    float* dst = out + (size_t)row * D + cA;
    dst[0]  = c0[r] + b0v;
    dst[16] = c1[r] + b1v;
    dst[32] = c2[r] + b2v;
    dst[48] = c3[r] + b3v;
  }
}

// ---------------------------------------------------------------------------
extern "C" void kernel_launch(void* const* d_in, const int* in_sizes, int n_in,
                              void* d_out, int out_size, void* d_ws,
                              size_t ws_size, hipStream_t stream) {
  const float* values  = (const float*)d_in[0];
  const float* queries = (const float*)d_in[1];
  const float* keys    = (const float*)d_in[2];
  const float* Wq = (const float*)d_in[3];
  const float* bq = (const float*)d_in[4];
  const float* Wk = (const float*)d_in[5];
  const float* bk = (const float*)d_in[6];
  const float* Wv = (const float*)d_in[7];
  const float* bv = (const float*)d_in[8];
  const float* Wo = (const float*)d_in[9];
  const float* bo = (const float*)d_in[10];

  constexpr int B = 4, S = 2048, H = 16, DK = 64, D = 1024;
  const size_t NQ = (size_t)B * H * S * DK;     // per-tensor bf16 elements
  unsigned short* ws  = (unsigned short*)d_ws;  // [q | k | v | ctx], bf16
  unsigned short* ctx = ws + 3 * NQ;
  float* out = (float*)d_out;

  qkv_proj_kernel<<<dim3(S / 128, B * H, 3), 256, 0, stream>>>(
      queries, keys, values, Wq, bq, Wk, bk, Wv, bv, ws);
  flash_attn_kernel<<<dim3(S / 128, B * H), 256, 0, stream>>>(ws, ctx);
  out_proj_kernel<<<dim3((B * S) / 128, D / 64), 256, 0, stream>>>(ctx, Wo, bo, out);
}
